// LSTM_gat_50843822850095
// MI455X (gfx1250) — matrix operand order repaired and checked by reading, not compiled, not run on hardware
//
#include <hip/hip_runtime.h>
#include <hip/hip_bf16.h>

typedef __attribute__((ext_vector_type(16))) _Float16 v16h;
typedef __attribute__((ext_vector_type(8)))  float    v8f;
typedef int v4i_vs __attribute__((vector_size(16)));   // matches builtin param type

#define B_     8
#define T_     128
#define M_     8
#define KP_    16
#define KR_    17
#define N_     136          // M_*KR_
#define NPAD_  144          // padded row count (multiple of 16)
#define EMB_   64
#define HID_   128
#define GATES_ 512          // 4*HID_
#define ZSTR_  516          // padded z row stride (banks shift 32 between half-waves)
#define NEG_SLOPE_ 0.2f
#define EMAX_  512

// ---- gfx1250 async global<->LDS paths (guarded; fallback = sync copies) ----
#if defined(__has_builtin)
# if __has_builtin(__builtin_amdgcn_global_load_async_to_lds_b32)
#  define HAVE_ASYNC_LD 1
# endif
# if __has_builtin(__builtin_amdgcn_global_store_async_from_lds_b128)
#  define HAVE_ASYNC_ST 1
# endif
#endif

__device__ __forceinline__ void wait_async0(){
#if defined(__has_builtin)
# if __has_builtin(__builtin_amdgcn_s_wait_asynccnt)
  __builtin_amdgcn_s_wait_asynccnt(0);
  return;
# endif
#endif
  asm volatile("s_wait_asynccnt 0x0" ::: "memory");
}

#ifdef HAVE_ASYNC_LD
__device__ __forceinline__ void async_g2l_b32(const void* g, void* l){
  __builtin_amdgcn_global_load_async_to_lds_b32(
      (__attribute__((address_space(1))) int*)g,
      (__attribute__((address_space(3))) int*)l, 0, 0);
}
#endif
#ifdef HAVE_ASYNC_ST
__device__ __forceinline__ void async_l2g_b128(void* g, const void* l){
  __builtin_amdgcn_global_store_async_from_lds_b128(
      (__attribute__((address_space(1))) v4i_vs*)g,
      (__attribute__((address_space(3))) v4i_vs*)l, 0, 0);
}
#endif

__device__ __forceinline__ float sigmoidf_(float x){ return 1.f/(1.f+__expf(-x)); }

// ---- WMMA fragment gathers (wave32, contiguous k order shared by A and B) ----

// A matrix 16x32 f16: lane L<16 -> row L, K = kb..kb+15 with kb=0;
// lane L>=16 -> row L-16, kb=16.  (Contiguous 32B -> 2x ds_load_b128.)
__device__ __forceinline__ v16h ldsA(const _Float16* base, int stride, int k0){
  int lane = threadIdx.x & 31;
  int row  = lane & 15;
  int kb   = (lane < 16) ? 0 : 16;
  v16h a;
#pragma unroll
  for (int h = 0; h < 16; ++h){
    int k = kb + h;
    a[h] = base[row * stride + k0 + k];
  }
  return a;
}

// B matrix 32x16 f16 (K x N): lanes 0-15 hold K=k0..k0+15 across 16 halves,
// lanes 16-31 hold K=k0+16..k0+31; N = n0 + lane%16.
__device__ __forceinline__ v16h ldsB(const _Float16* base, int stride, int k0, int n0){
  int lane = threadIdx.x & 31;
  int n    = n0 + (lane & 15);
  int kb   = (lane < 16) ? 0 : 16;
  v16h b;
#pragma unroll
  for (int h = 0; h < 16; ++h)
    b[h] = base[(k0 + kb + h) * stride + n];
  return b;
}

// B fragment gathered from a row-major (OUT x IN) f32 weight, i.e. element(k,n)=W[n*ldw+k].
__device__ __forceinline__ v16h gblBT(const float* W, int ldw, int k0, int n0){
  int lane = threadIdx.x & 31;
  int n    = n0 + (lane & 15);
  int kb   = (lane < 16) ? 0 : 16;
  v16h b;
#pragma unroll
  for (int h = 0; h < 16; ++h)
    b[h] = (_Float16)W[n * ldw + k0 + kb + h];
  return b;
}

// =====================================================================
// Kernel 1: root keypoint + MLP (3 -> 64 relu -> 64) ; v written as f16
// into a row-padded [bt][144][64] workspace.
// =====================================================================
__global__ __launch_bounds__(256)
void k1_feat_mlp(const float* __restrict__ feat, const float* __restrict__ W1,
                 const float* __restrict__ b1, const float* __restrict__ W2,
                 const float* __restrict__ b2, _Float16* __restrict__ v_out){
  __shared__ float     rootf[M_][3];
  __shared__ float     featR[N_][3];
  __shared__ __align__(16) _Float16  h1[NPAD_ * EMB_];
  __shared__ __align__(16) _Float16  w2s[EMB_ * EMB_];
  __shared__ __align__(16) _Float16  vstage[NPAD_ * EMB_];
  int tid = threadIdx.x;
  int bt  = blockIdx.x;
  const float* f0 = feat + (size_t)bt * (M_ * KP_ * 3);

  if (tid < M_){
    float sx = 0.f, sy = 0.f, sw = 0.f; int any = 0;
    for (int k = 0; k < KP_; ++k){
      const float* p = f0 + (tid * KP_ + k) * 3;
      float w = (p[2] > 0.5f) ? 1.f : 0.f;
      sx += p[0] * w; sy += p[1] * w; sw += w;
      any |= (p[2] > 0.5f);
    }
    float den = sw < 1.f ? 1.f : sw;
    rootf[tid][0] = sx / den; rootf[tid][1] = sy / den; rootf[tid][2] = any ? 1.f : 0.f;
  }
  for (int i = tid; i < EMB_ * EMB_; i += 256) w2s[i] = (_Float16)W2[i];
  __syncthreads();

  for (int n = tid; n < N_; n += 256){
    int m = n / KR_, kp = n % KR_;
    if (kp < KP_){
      const float* p = f0 + (m * KP_ + kp) * 3;
      featR[n][0] = p[0]; featR[n][1] = p[1]; featR[n][2] = p[2];
    } else {
      featR[n][0] = rootf[m][0]; featR[n][1] = rootf[m][1]; featR[n][2] = rootf[m][2];
    }
  }
  __syncthreads();

  for (int i = tid; i < NPAD_ * EMB_; i += 256){
    int n = i >> 6, j = i & 63;
    float v = 0.f;
    if (n < N_){
      v = featR[n][0] * W1[j] + featR[n][1] * W1[EMB_ + j] +
          featR[n][2] * W1[2 * EMB_ + j] + b1[j];
      v = v > 0.f ? v : 0.f;
    }
    h1[i] = (_Float16)v;
  }
  __syncthreads();

  int wave = tid >> 5, lane = tid & 31;
  for (int tt = wave; tt < 36; tt += 8){
    int rt = tt >> 2, nt = tt & 3;
    int col = nt * 16 + (lane & 15);
    float bc = b2[col];
    v8f acc;
#pragma unroll
    for (int r = 0; r < 8; ++r) acc[r] = bc;
#pragma unroll
    for (int kc = 0; kc < 2; ++kc){
      v16h a = ldsA(h1 + rt * 16 * EMB_, EMB_, kc * 32);
      v16h b = ldsB(w2s, EMB_, kc * 32, nt * 16);
      acc = __builtin_amdgcn_wmma_f32_16x16x32_f16(false, a, false, b, (short)0, acc, false, false);
    }
#pragma unroll
    for (int r = 0; r < 8; ++r){
      int row = rt * 16 + r + ((lane < 16) ? 0 : 8);
      vstage[row * EMB_ + col] = (_Float16)acc[r];
    }
  }
  __syncthreads();
  // whole-tile LDS -> global copy (async DMA path when available)
  {
    const uint4* s4 = (const uint4*)vstage;
    uint4* g4 = (uint4*)(v_out + (size_t)bt * (NPAD_ * EMB_));
#ifdef HAVE_ASYNC_ST
    for (int i = tid; i < (NPAD_ * EMB_) / 8; i += 256)
      async_l2g_b128(g4 + i, s4 + i);
    wait_async0();
#else
    for (int i = tid; i < (NPAD_ * EMB_) / 8; i += 256) g4[i] = s4[i];
#endif
  }
}

// =====================================================================
// Kernel 2: LSTM over T.  16 rows per block, 16 waves; wave w owns gate
// columns [32w, 32w+32). Weight B-fragments live in registers for the
// whole T loop; h/c state lives in LDS.  x tiles double-buffered with
// async global->LDS DMA; dual independent WMMA accumulator chains.
// =====================================================================
__global__ __launch_bounds__(512)
void k2_lstm(const _Float16* __restrict__ v_in, const float* __restrict__ W_ih,
             const float* __restrict__ W_hh, const float* __restrict__ b_ih,
             const float* __restrict__ b_hh, float* __restrict__ r_out){
  __shared__ __align__(16) _Float16  xs[2][16 * EMB_];
  __shared__ __align__(16) _Float16  hs[16 * HID_];
  __shared__ float     cs[16 * HID_];
  __shared__ float     zs[16 * ZSTR_];
  __shared__ long      vbase[16];     // in halves, even
  __shared__ long      rbase[16];

  int tid  = threadIdx.x;
  int lane = tid & 31;
  int wave = tid >> 5;            // 0..15
  int rb0  = blockIdx.x * 16;

  if (tid < 16){
    int rb = rb0 + tid;
    int bb = rb / N_, nn = rb % N_;
    vbase[tid] = ((long)bb * T_ * NPAD_ + nn) * EMB_;
    rbase[tid] = ((long)bb * T_ * N_ + nn) * HID_;
  }
  for (int i = tid; i < 16 * HID_; i += 512){ hs[i] = (_Float16)0.f; cs[i] = 0.f; }

  int n0w = wave * 32;
  float bcol0 = b_ih[n0w + (lane & 15)]      + b_hh[n0w + (lane & 15)];
  float bcol1 = b_ih[n0w + 16 + (lane & 15)] + b_hh[n0w + 16 + (lane & 15)];

  // preload time-invariant B fragments into registers
  v16h Bx[2][2], Bh[4][2];
#pragma unroll
  for (int ti = 0; ti < 2; ++ti){
#pragma unroll
    for (int kc = 0; kc < 2; ++kc) Bx[kc][ti] = gblBT(W_ih, EMB_, kc * 32, n0w + ti * 16);
#pragma unroll
    for (int kc = 0; kc < 4; ++kc) Bh[kc][ti] = gblBT(W_hh, HID_, kc * 32, n0w + ti * 16);
  }
  __syncthreads();   // vbase visible

  int r  = tid >> 5, kp = tid & 31;               // each wave stages one row
  const unsigned* vsrc = (const unsigned*)v_in;
  long vrow = (vbase[r] >> 1) + kp;               // u32 index, step NPAD_*EMB_/2 per t

#ifdef HAVE_ASYNC_LD
  async_g2l_b32(vsrc + vrow, (unsigned*)xs[0] + tid);   // t = 0
  wait_async0();
#else
  ((unsigned*)xs[0])[tid] = vsrc[vrow];
#endif
  __syncthreads();

  for (int t = 0; t < T_; ++t){
    const _Float16* xcur = xs[t & 1];
    if (t + 1 < T_){
#ifdef HAVE_ASYNC_LD
      async_g2l_b32(vsrc + vrow + (long)(t + 1) * (NPAD_ * EMB_ / 2),
                    (unsigned*)xs[(t + 1) & 1] + tid);
#else
      ((unsigned*)xs[(t + 1) & 1])[tid] = vsrc[vrow + (long)(t + 1) * (NPAD_ * EMB_ / 2)];
#endif
    }

    v16h Ax[2], Ah[4];
#pragma unroll
    for (int kc = 0; kc < 2; ++kc) Ax[kc] = ldsA(xcur, EMB_, kc * 32);
#pragma unroll
    for (int kc = 0; kc < 4; ++kc) Ah[kc] = ldsA(hs, HID_, kc * 32);

    // two independent accumulator chains, interleaved to fill WMMA hazard slots
    v8f acc0, acc1;
#pragma unroll
    for (int q = 0; q < 8; ++q){ acc0[q] = bcol0; acc1[q] = bcol1; }
#pragma unroll
    for (int kc = 0; kc < 2; ++kc){
      acc0 = __builtin_amdgcn_wmma_f32_16x16x32_f16(false, Ax[kc], false, Bx[kc][0], (short)0, acc0, false, false);
      acc1 = __builtin_amdgcn_wmma_f32_16x16x32_f16(false, Ax[kc], false, Bx[kc][1], (short)0, acc1, false, false);
    }
#pragma unroll
    for (int kc = 0; kc < 4; ++kc){
      acc0 = __builtin_amdgcn_wmma_f32_16x16x32_f16(false, Ah[kc], false, Bh[kc][0], (short)0, acc0, false, false);
      acc1 = __builtin_amdgcn_wmma_f32_16x16x32_f16(false, Ah[kc], false, Bh[kc][1], (short)0, acc1, false, false);
    }
    {
      int col0 = n0w + (lane & 15);
      int rbase2 = (lane < 16) ? 0 : 8;
#pragma unroll
      for (int q = 0; q < 8; ++q){
        int row = q + rbase2;
        zs[row * ZSTR_ + col0]      = acc0[q];
        zs[row * ZSTR_ + col0 + 16] = acc1[q];
      }
    }
    __syncthreads();

    // gate nonlinearities + state update + stream h to r (bias already in z)
    for (int i = tid; i < 16 * HID_; i += 512){
      int rr = i >> 7, j = i & 127;
      float zi = zs[rr * ZSTR_ + j];
      float zf = zs[rr * ZSTR_ + HID_ + j];
      float zg = zs[rr * ZSTR_ + 2 * HID_ + j];
      float zo = zs[rr * ZSTR_ + 3 * HID_ + j];
      float c = sigmoidf_(zf) * cs[i] + sigmoidf_(zi) * tanhf(zg);
      float h = sigmoidf_(zo) * tanhf(c);
      cs[i] = c;
      hs[i] = (_Float16)h;
      r_out[rbase[rr] + (long)t * (N_ * HID_) + j] = h;
    }
#ifdef HAVE_ASYNC_LD
    wait_async0();       // t+1 x tile landed (this wave's transfers)
#endif
    __syncthreads();     // all waves' tiles + h/c visible
  }
}

// =====================================================================
// Kernel 3: fused h = r @ Wg (WMMA) + edge-softmax GAT + ELU, per (b,t)
// Row-tiles processed in pairs -> two independent WMMA chains.
// =====================================================================
__global__ __launch_bounds__(256)
void k3_gat(const float* __restrict__ r_in, const float* __restrict__ Wg,
            const float* __restrict__ a_src, const float* __restrict__ a_dst,
            const int* __restrict__ srcE, const int* __restrict__ dstE,
            int E, float* __restrict__ rhat_out){
  extern __shared__ unsigned char smem[];
  _Float16* rstage = (_Float16*)smem;                 // 144*128 f16
  _Float16* wgs    = rstage + NPAD_ * HID_;           // 128*128 f16
  _Float16* hbuf   = wgs + HID_ * HID_;               // 144*128 f16
  float* ssrc = (float*)(hbuf + NPAD_ * HID_);        // 144
  float* sdst = ssrc + NPAD_;                         // 144
  float* mmax = sdst + NPAD_;                         // 144
  float* den  = mmax + NPAD_;                         // 144
  float* alph = den + NPAD_;                          // EMAX_
  int*   es   = (int*)(alph + EMAX_);                 // EMAX_
  int*   ed   = es + EMAX_;                           // EMAX_

  int tid = threadIdx.x;
  int bt  = blockIdx.x;
  const float* rbt = r_in + (size_t)bt * (N_ * HID_);

  for (int i = tid; i < NPAD_ * HID_; i += 256){
    int n = i >> 7;
    rstage[i] = (n < N_) ? (_Float16)rbt[i] : (_Float16)0.f;
  }
  for (int i = tid; i < HID_ * HID_; i += 256) wgs[i] = (_Float16)Wg[i];
  for (int e = tid; e < E; e += 256){ es[e] = srcE[e]; ed[e] = dstE[e]; }
  __syncthreads();

  int wave = tid >> 5, lane = tid & 31;
  int n0 = wave * 16;
  int col = n0 + (lane & 15);
  int rofs = (lane < 16) ? 0 : 8;
  // 9 row tiles: 4 pairs + 1 tail ; pairs give two independent WMMA chains
  for (int rtp = 0; rtp < 4; ++rtp){
    int rt0 = rtp * 2, rt1 = rt0 + 1;
    v8f acc0 = {}, acc1 = {};
#pragma unroll
    for (int kc = 0; kc < 4; ++kc){
      v16h b  = ldsB(wgs, HID_, kc * 32, n0);
      v16h a0 = ldsA(rstage + rt0 * 16 * HID_, HID_, kc * 32);
      v16h a1 = ldsA(rstage + rt1 * 16 * HID_, HID_, kc * 32);
      acc0 = __builtin_amdgcn_wmma_f32_16x16x32_f16(false, a0, false, b, (short)0, acc0, false, false);
      acc1 = __builtin_amdgcn_wmma_f32_16x16x32_f16(false, a1, false, b, (short)0, acc1, false, false);
    }
#pragma unroll
    for (int q = 0; q < 8; ++q){
      hbuf[(rt0 * 16 + q + rofs) * HID_ + col] = (_Float16)acc0[q];
      hbuf[(rt1 * 16 + q + rofs) * HID_ + col] = (_Float16)acc1[q];
    }
  }
  {
    v8f acc = {};
#pragma unroll
    for (int kc = 0; kc < 4; ++kc){
      v16h a = ldsA(rstage + 8 * 16 * HID_, HID_, kc * 32);
      v16h b = ldsB(wgs, HID_, kc * 32, n0);
      acc = __builtin_amdgcn_wmma_f32_16x16x32_f16(false, a, false, b, (short)0, acc, false, false);
    }
#pragma unroll
    for (int q = 0; q < 8; ++q)
      hbuf[(8 * 16 + q + rofs) * HID_ + col] = (_Float16)acc[q];
  }
  __syncthreads();

  for (int n = tid; n < N_; n += 256){
    float s1 = 0.f, s2 = 0.f;
    for (int j = 0; j < HID_; ++j){
      float hv = (float)hbuf[n * HID_ + j];
      s1 += hv * a_src[j];
      s2 += hv * a_dst[j];
    }
    ssrc[n] = s1; sdst[n] = s2;
  }
  __syncthreads();

  for (int d = tid; d < N_; d += 256){
    float mm = -3.0e38f;
    for (int e = 0; e < E; ++e) if (ed[e] == d){
      float ev = ssrc[es[e]] + sdst[d];
      ev = ev > 0.f ? ev : NEG_SLOPE_ * ev;
      mm = mm > ev ? mm : ev;
    }
    mmax[d] = mm;
  }
  __syncthreads();

  for (int e = tid; e < E; e += 256){
    float ev = ssrc[es[e]] + sdst[ed[e]];
    ev = ev > 0.f ? ev : NEG_SLOPE_ * ev;
    alph[e] = __expf(ev - mmax[ed[e]]);
  }
  __syncthreads();

  for (int d = tid; d < N_; d += 256){
    float s = 0.f;
    for (int e = 0; e < E; ++e) if (ed[e] == d) s += alph[e];
    den[d] = (s > 0.f) ? s : 1.f;
  }
  __syncthreads();

  for (int e = tid; e < E; e += 256) alph[e] /= den[ed[e]];
  __syncthreads();

  float* out = rhat_out + (size_t)bt * (N_ * HID_);
  for (int i = tid; i < N_ * HID_; i += 256){
    int d = i >> 7, j = i & 127;
    float s = 0.f;
    for (int e = 0; e < E; ++e)
      if (ed[e] == d) s += alph[e] * (float)hbuf[es[e] * HID_ + j];
    out[i] = s > 0.f ? s : (__expf(s) - 1.f);
  }
}

// =====================================================================
extern "C" void kernel_launch(void* const* d_in, const int* in_sizes, int n_in,
                              void* d_out, int out_size, void* d_ws, size_t ws_size,
                              hipStream_t stream){
  (void)n_in; (void)out_size; (void)ws_size;
  const float* feat  = (const float*)d_in[0];
  const float* W1    = (const float*)d_in[1];
  const float* b1    = (const float*)d_in[2];
  const float* W2    = (const float*)d_in[3];
  const float* b2    = (const float*)d_in[4];
  const float* W_ih  = (const float*)d_in[5];
  const float* W_hh  = (const float*)d_in[6];
  const float* b_ih  = (const float*)d_in[7];
  const float* b_hh  = (const float*)d_in[8];
  const float* Wg    = (const float*)d_in[9];
  const float* a_src = (const float*)d_in[10];
  const float* a_dst = (const float*)d_in[11];
  const int*   srcE  = (const int*)d_in[12];
  const int*   dstE  = (const int*)d_in[13];
  int E = in_sizes[12]; if (E > EMAX_) E = EMAX_;

  float* r_out    = (float*)d_out;
  float* rhat_out = r_out + (size_t)B_ * T_ * N_ * HID_;
  _Float16* v_ws  = (_Float16*)d_ws;    // B*T*144*EMB f16 = ~18.9 MB (row-padded)

  hipLaunchKernelGGL(k1_feat_mlp, dim3(B_ * T_), dim3(256), 0, stream,
                     feat, W1, b1, W2, b2, v_ws);
  hipLaunchKernelGGL(k2_lstm, dim3((B_ * N_) / 16), dim3(512), 0, stream,
                     v_ws, W_ih, W_hh, b_ih, b_hh, r_out);

  size_t sm3 = (size_t)(NPAD_ * HID_) * 2     // rstage
             + (size_t)(HID_ * HID_) * 2      // wgs
             + (size_t)(NPAD_ * HID_) * 2     // hbuf
             + 4 * NPAD_ * sizeof(float)      // ssrc, sdst, mmax, den
             + EMAX_ * sizeof(float)          // alph
             + 2 * EMAX_ * sizeof(int);       // es, ed
  hipLaunchKernelGGL(k3_gat, dim3(B_ * T_), dim3(256), sm3, stream,
                     r_out, Wg, a_src, a_dst, srcE, dstE, E, rhat_out);
}
